// AttentionLayer_146028888352
// MI455X (gfx1250) — compile-verified, matrix operand order checked by reading, not run on hardware
//
#include <hip/hip_runtime.h>
#include <cstddef>
#include <cstdint>

// ---------------------------------------------------------------------------
// Problem constants (match reference): B=4, L=S=2048, D=512, H=8, DH=64
// ---------------------------------------------------------------------------
#define BB 4
#define LL 2048
#define SS 2048
#define DD 512
#define HH 8
#define DH 64

typedef __attribute__((ext_vector_type(16))) _Float16 v16h;
typedef __attribute__((ext_vector_type(8)))  float    v8f;

#ifndef __has_builtin
#define __has_builtin(x) 0
#endif

#if __has_builtin(__builtin_amdgcn_tensor_load_to_lds) && \
    __has_builtin(__builtin_amdgcn_s_wait_tensorcnt)
#define USE_TDM 1
#else
#define USE_TDM 0
#endif

__device__ __forceinline__ v8f v8f_zero() {
  v8f z = {0.f, 0.f, 0.f, 0.f, 0.f, 0.f, 0.f, 0.f};
  return z;
}

// CDNA5 WMMA f32 = f16 x f16 + f32, 16x16x32, wave32.
__device__ __forceinline__ v8f wmma_f16(v16h a, v16h b, v8f c) {
  return __builtin_amdgcn_wmma_f32_16x16x32_f16(
      /*neg_a=*/false, a, /*neg_b=*/false, b,
      /*c_mod=*/(short)0, c, /*reuse_a=*/false, /*reuse_b=*/false);
}

// 16-bit A/B fragment K index for half-slot h (0..15) in a given lane.
// CDNA5 ISA: lanes 0-15 hold K=0..7 (VGPR0-3) and K=16..23 (VGPR4-7);
// lanes 16-31 hold K=8..15 and K=24..31.
__device__ __forceinline__ int frag_k(int h, int lane) {
  return h + 8 * ((h >> 3) + (lane >> 4));
}

// ---------------------------------------------------------------------------
// Tensor Data Mover: DMA a [rows x 64] f16 tile (global row stride
// `stride_elems` halfs) into contiguous LDS at `lds_dst`. Wave-uniform args;
// issue from a single wave, then s_wait_tensorcnt(0) before the barrier.
// D# layout per CDNA5 ISA.
// ---------------------------------------------------------------------------
#if USE_TDM
typedef __attribute__((ext_vector_type(4))) unsigned int u32x4;
typedef __attribute__((ext_vector_type(4))) int          i32x4;
typedef __attribute__((ext_vector_type(8))) int          i32x8;

__device__ __forceinline__ void tdm_load_rows_f16(void* lds_dst,
                                                  const void* gsrc, int rows,
                                                  int stride_elems) {
  const uint64_t ga = (uint64_t)(uintptr_t)gsrc;
  u32x4 g0;
  g0.x = 1u;                                  // count=1, user descriptor
  g0.y = (uint32_t)(uintptr_t)lds_dst;        // lds_addr (flat addr[31:0])
  g0.z = (uint32_t)ga;                        // global_addr[31:0]
  g0.w = (uint32_t)((ga >> 32) & 0x1FFFFFFu)  // global_addr[56:32]
         | (2u << 30);                        // type = 2 ("image")
  i32x8 g1;
  g1[0] = 0x10000;                      // workgroup_mask=0, data_size=1 (2B)
  g1[1] = (int)(64u << 16);             // tensor_dim0 = 64 (bits 79:48 lo16)
  g1[2] = (int)((unsigned)rows << 16);  // tensor_dim1 = rows (bits 111:80 lo16)
  g1[3] = (int)(64u << 16);             // tile_dim0 = 64   (bits 127:112)
  g1[4] = rows;                         // tile_dim1 = rows (bits 143:128)
  g1[5] = stride_elems;                 // tensor_dim0_stride[31:0]
  g1[6] = 0;                            // stride hi / dim1_stride lo
  g1[7] = 0;
  i32x4 z4 = {0, 0, 0, 0};
#if defined(__clang_major__) && __clang_major__ >= 23
  i32x8 z8 = {0, 0, 0, 0, 0, 0, 0, 0};
  __builtin_amdgcn_tensor_load_to_lds(g0, g1, z4, z4, z8, 0);
#else
  __builtin_amdgcn_tensor_load_to_lds(g0, g1, z4, z4, 0);
#endif
}
#endif  // USE_TDM

// Stage a [rows x 64]-f16 tile into LDS; caller must __syncthreads() after.
__device__ __forceinline__ void stage_tile_f16(_Float16* dst,
                                               const _Float16* src, int rows,
                                               int stride_elems, int tid,
                                               int wid) {
#if USE_TDM
  if (wid == 0) {
    tdm_load_rows_f16((void*)dst, (const void*)src, rows, stride_elems);
    __builtin_amdgcn_s_wait_tensorcnt(0);
  }
#else
  for (int idx = tid; idx < rows * 64; idx += 256)
    dst[idx] = src[(size_t)(idx >> 6) * stride_elems + (idx & 63)];
#endif
}

// ---------------------------------------------------------------------------
// GEMM: C[M,N] = X[M,K] @ W[N,K]^T + bias[N]
// Block tile 128x64, BK=32, 256 threads (8 waves). Wave w computes rows
// [w*16, w*16+16) x 64 cols (4 WMMA N-subtiles). X staged to LDS as f16.
// XF16: X is _Float16 (else float). OUTF16: C stored _Float16 (else float).
// ---------------------------------------------------------------------------
template <bool XF16, bool OUTF16>
__global__ __launch_bounds__(256) void gemm_bias_kernel(
    const void* __restrict__ Xv, const float* __restrict__ W,
    const float* __restrict__ bias, void* __restrict__ Cv,
    int M, int N, int K) {
  __shared__ _Float16 a_lds[128 * 32];
  __shared__ _Float16 w_lds[64 * 32];

  const int tid  = threadIdx.x;
  const int lane = tid & 31;
  const int wid  = tid >> 5;
  const int m0   = blockIdx.x * 128;
  const int n0   = blockIdx.y * 64;
  const int mrow = lane & 15;

  v8f acc[4];
#pragma unroll
  for (int nt = 0; nt < 4; ++nt) acc[nt] = v8f_zero();

  for (int k0 = 0; k0 < K; k0 += 32) {
    // Prefetch next K-tile of W and X into cache while this tile streams.
    if (k0 + 32 < K) {
      __builtin_prefetch(&W[(size_t)(n0 + (tid & 63)) * K + k0 + 32], 0, 3);
      if (XF16)
        __builtin_prefetch(
            &((const _Float16*)Xv)[(size_t)(m0 + (tid & 127)) * K + k0 + 32],
            0, 3);
      else
        __builtin_prefetch(
            &((const float*)Xv)[(size_t)(m0 + (tid & 127)) * K + k0 + 32], 0,
            3);
    }
    // Stage X tile [128 x 32] -> f16 LDS
    for (int idx = tid; idx < 128 * 32; idx += 256) {
      const int i = idx >> 5, j = idx & 31;
      float xv;
      if (XF16)
        xv = (float)((const _Float16*)Xv)[(size_t)(m0 + i) * K + k0 + j];
      else
        xv = ((const float*)Xv)[(size_t)(m0 + i) * K + k0 + j];
      a_lds[idx] = (_Float16)xv;
    }
    // Stage W tile [64 x 32] -> f16 LDS (W is [N,K] row-major)
    for (int idx = tid; idx < 64 * 32; idx += 256) {
      const int i = idx >> 5, j = idx & 31;
      w_lds[idx] = (_Float16)W[(size_t)(n0 + i) * K + k0 + j];
    }
    __syncthreads();

    // A fragment: rows wid*16 .. wid*16+15
    v16h afr;
#pragma unroll
    for (int h = 0; h < 16; ++h)
      afr[h] = a_lds[(wid * 16 + mrow) * 32 + frag_k(h, lane)];

#pragma unroll
    for (int nt = 0; nt < 4; ++nt) {
      v16h bfr;  // B = W^T: B[k][n] = W[n][k] = w_lds[n][k]
#pragma unroll
      for (int h = 0; h < 16; ++h)
        bfr[h] = w_lds[(nt * 16 + mrow) * 32 + frag_k(h, lane)];
      acc[nt] = wmma_f16(afr, bfr, acc[nt]);
    }
    __syncthreads();
  }

  // C layout: VGPR r -> M = r + 8*(lane>=16); N = lane%16
#pragma unroll
  for (int nt = 0; nt < 4; ++nt) {
#pragma unroll
    for (int r = 0; r < 8; ++r) {
      const int m = wid * 16 + r + ((lane >> 4) << 3);
      const int n = nt * 16 + mrow;
      const float v = acc[nt][r] + bias[n0 + n];
      if (OUTF16)
        ((_Float16*)Cv)[(size_t)(m0 + m) * N + n0 + n] = (_Float16)v;
      else
        ((float*)Cv)[(size_t)(m0 + m) * N + n0 + n] = v;
    }
  }
}

// ---------------------------------------------------------------------------
// Attention: one workgroup per (b, h, 16-query tile). 256 threads (8 waves).
// Dynamic LDS layout (171072 B total, fits 320 KB/WGP):
//   scores  : 16 * 2048 f32  = 131072 B
//   kvstage : 256 * 64  f16  =  32768 B   (TDM destination)
//   qtile   : 16  * 64  f16  =   2048 B   (TDM destination)
//   red     : 16  * 16  f32  =   1024 B
//   rowstat : 16        f32  =     64 B
//   out_acc : 16  * 64  f32  =   4096 B
// ---------------------------------------------------------------------------
#define ATTN_SMEM_BYTES (131072 + 32768 + 2048 + 1024 + 64 + 4096)

__global__ __launch_bounds__(256) void attn_kernel(
    const _Float16* __restrict__ Qb, const _Float16* __restrict__ Kb,
    const _Float16* __restrict__ Vb, float* __restrict__ attn,
    _Float16* __restrict__ ctx) {
  extern __shared__ char smem[];
  float*    scores  = (float*)smem;                               // [16][2048]
  _Float16* kvstage = (_Float16*)(smem + 131072);                 // [256][64]
  _Float16* qtile   = (_Float16*)(smem + 131072 + 32768);         // [16][64]
  float*    red     = (float*)(smem + 131072 + 32768 + 2048);     // [16][16]
  float*    rowstat = (float*)(smem + 131072 + 32768 + 2048 + 1024);
  float*    out_acc = (float*)(smem + 131072 + 32768 + 2048 + 1024 + 64);

  const int tid  = threadIdx.x;
  const int lane = tid & 31;
  const int wid  = tid >> 5;
  const int mrow = lane & 15;

  const int wg = blockIdx.x;
  const int qt = wg & (LL / 16 - 1);      // 0..127
  const int bh = wg >> 7;                 // 0..31
  const int bi = bh >> 3;                 // batch
  const int hd = bh & 7;                  // head
  const int q0 = qt * 16;

  // ---- Stage Q tile [16 x 64] (f16) via TDM ----
  stage_tile_f16(qtile, Qb + ((size_t)bi * LL + q0) * DD + hd * DH, 16, DD,
                 tid, wid);
  __syncthreads();

  // Preload Q A-fragments (K = DH = 64 -> two 32-wide WMMA steps)
  v16h aq[2];
#pragma unroll
  for (int ks = 0; ks < 2; ++ks)
#pragma unroll
    for (int h = 0; h < 16; ++h)
      aq[ks][h] = qtile[mrow * DH + ks * 32 + frag_k(h, lane)];

  // ---- Scores = scale * Q @ K^T, chunked over S (256 keys/chunk) ----
  const float scale = 0.125f;  // 1/sqrt(64)
  for (int ch = 0; ch < SS / 256; ++ch) {
    const int s0 = ch * 256;
    stage_tile_f16(kvstage, Kb + ((size_t)bi * SS + s0) * DD + hd * DH, 256,
                   DD, tid, wid);
    __syncthreads();

#pragma unroll
    for (int nt = 0; nt < 2; ++nt) {      // wave owns 32 key-columns
      const int ncol = wid * 32 + nt * 16;
      v8f acc = v8f_zero();
#pragma unroll
      for (int ks = 0; ks < 2; ++ks) {
        v16h bfr;  // B[k=dh][n=key] = kvstage[key][dh]
#pragma unroll
        for (int h = 0; h < 16; ++h)
          bfr[h] = kvstage[(ncol + mrow) * DH + ks * 32 + frag_k(h, lane)];
        acc = wmma_f16(aq[ks], bfr, acc);
      }
#pragma unroll
      for (int r = 0; r < 8; ++r) {
        const int mm = r + ((lane >> 4) << 3);
        scores[mm * SS + s0 + ncol + mrow] = acc[r] * scale;
      }
    }
    __syncthreads();
  }

  // ---- Softmax over S (16 threads per row, 128 elements each) ----
  const int row = tid >> 4, sub = tid & 15;
  float* srow = scores + row * SS + sub * 128;
  float mx = -3.4e38f;
#pragma unroll 4
  for (int j = 0; j < 128; ++j) mx = fmaxf(mx, srow[j]);
  red[row * 16 + sub] = mx;
  __syncthreads();
  float rowmax = red[row * 16];
#pragma unroll
  for (int i = 1; i < 16; ++i) rowmax = fmaxf(rowmax, red[row * 16 + i]);
  __syncthreads();  // all reads of max partials done before reuse

  float sum = 0.f;
#pragma unroll 4
  for (int j = 0; j < 128; ++j) {
    const float e = __expf(srow[j] - rowmax);
    srow[j] = e;
    sum += e;
  }
  red[row * 16 + sub] = sum;
  __syncthreads();
  float total = 0.f;
#pragma unroll
  for (int i = 0; i < 16; ++i) total += red[row * 16 + i];
  if (sub == 0) rowstat[row] = 1.0f / total;
  __syncthreads();

  // ---- Write normalized attn [B,H,L,S] (coalesced along S) ----
  const size_t abase = ((size_t)(bi * HH + hd) * LL + q0) * SS;
  for (int idx = tid; idx < 16 * SS; idx += 256) {
    const int rr = idx >> 11;
    attn[abase + (size_t)rr * SS + (idx & (SS - 1))] =
        scores[idx] * rowstat[rr];
  }

  // ---- PV: out[16x64] = attn @ V, inv(sum) folded into A-gather ----
  for (int idx = tid; idx < 16 * DH; idx += 256) out_acc[idx] = 0.f;
  __syncthreads();

  const float rinv = rowstat[mrow];
  v8f pv[4];
#pragma unroll
  for (int nt = 0; nt < 4; ++nt) pv[nt] = v8f_zero();

  for (int ch = 0; ch < SS / 256; ++ch) {
    const int s0 = ch * 256;
    stage_tile_f16(kvstage, Vb + ((size_t)bi * SS + s0) * DD + hd * DH, 256,
                   DD, tid, wid);
    __syncthreads();

    // Wave wid contracts keys [s0 + wid*32, s0 + wid*32 + 32)
    v16h afr;
#pragma unroll
    for (int h = 0; h < 16; ++h)
      afr[h] = (_Float16)(scores[mrow * SS + s0 + wid * 32 + frag_k(h, lane)] *
                          rinv);
#pragma unroll
    for (int nt = 0; nt < 4; ++nt) {
      v16h bfr;  // B[k=key][n=dh] = kvstage[key][dh]
#pragma unroll
      for (int h = 0; h < 16; ++h)
        bfr[h] = kvstage[(wid * 32 + frag_k(h, lane)) * DH + nt * 16 + mrow];
      pv[nt] = wmma_f16(afr, bfr, pv[nt]);
    }
    __syncthreads();
  }

  // Cross-wave reduction of partial PV sums via LDS float atomics
#pragma unroll
  for (int nt = 0; nt < 4; ++nt)
#pragma unroll
    for (int r = 0; r < 8; ++r) {
      const int mm = r + ((lane >> 4) << 3);
      const int nn = nt * 16 + mrow;
      atomicAdd(&out_acc[mm * DH + nn], pv[nt][r]);
    }
  __syncthreads();

  // ---- Store context (f16) for the output projection ----
  for (int idx = tid; idx < 16 * DH; idx += 256)
    ctx[((size_t)bi * LL + q0 + (idx >> 6)) * DD + hd * DH + (idx & 63)] =
        (_Float16)out_acc[idx];
}

// ---------------------------------------------------------------------------
// Host launcher
// ---------------------------------------------------------------------------
extern "C" void kernel_launch(void* const* d_in, const int* in_sizes, int n_in,
                              void* d_out, int out_size, void* d_ws,
                              size_t ws_size, hipStream_t stream) {
  (void)in_sizes; (void)n_in; (void)out_size; (void)ws_size;

  const float* queries = (const float*)d_in[0];  // [B,L,D]
  const float* keys    = (const float*)d_in[1];  // [B,S,D]
  const float* values  = (const float*)d_in[2];  // [B,S,D]
  const float* Wq = (const float*)d_in[3];
  const float* bq = (const float*)d_in[4];
  const float* Wk = (const float*)d_in[5];
  const float* bk = (const float*)d_in[6];
  const float* Wv = (const float*)d_in[7];
  const float* bv = (const float*)d_in[8];
  const float* Wo = (const float*)d_in[9];
  const float* bo = (const float*)d_in[10];

  float* out  = (float*)d_out;                         // [B,L,D]
  float* attn = out + (size_t)BB * LL * DD;            // [B,H,L,S]

  const size_t n_tok = (size_t)BB * LL * DD;           // 4,194,304 elems
  char* ws = (char*)d_ws;
  _Float16* Qb  = (_Float16*)ws;
  _Float16* Kb  = (_Float16*)(ws + n_tok * 2);
  _Float16* Vb  = (_Float16*)(ws + n_tok * 4);
  _Float16* ctx = (_Float16*)(ws + n_tok * 6);

  const int M = BB * LL;  // 8192
  const dim3 gblk(256);
  const dim3 ggrd(M / 128, DD / 64);  // (64, 8)

  // QKV projections: f32 in -> f16 out
  gemm_bias_kernel<false, true><<<ggrd, gblk, 0, stream>>>(
      (const void*)queries, Wq, bq, (void*)Qb, M, DD, DD);
  gemm_bias_kernel<false, true><<<ggrd, gblk, 0, stream>>>(
      (const void*)keys, Wk, bk, (void*)Kb, M, DD, DD);
  gemm_bias_kernel<false, true><<<ggrd, gblk, 0, stream>>>(
      (const void*)values, Wv, bv, (void*)Vb, M, DD, DD);

  // Attention: 4 * 8 * (2048/16) = 4096 workgroups, 171 KB LDS each
  static_assert(ATTN_SMEM_BYTES <= 320 * 1024, "LDS over WGP budget");
  hipFuncSetAttribute((const void*)attn_kernel,
                      hipFuncAttributeMaxDynamicSharedMemorySize,
                      ATTN_SMEM_BYTES);
  attn_kernel<<<dim3(BB * HH * (LL / 16)), gblk, ATTN_SMEM_BYTES, stream>>>(
      Qb, Kb, Vb, attn, ctx);

  // Output projection: f16 ctx -> f32 out
  gemm_bias_kernel<true, false><<<ggrd, gblk, 0, stream>>>(
      (const void*)ctx, Wo, bo, (void*)out, M, DD, DD);
}
